// TSCH_NN_70403103916225
// MI455X (gfx1250) — matrix operand order repaired and checked by reading, not compiled, not run on hardware
//
#include <hip/hip_runtime.h>

#define N_NODES    20000   // == 1250 * 16, all row tiles are full
#define N_EDGES    320000
#define H_HEADS    16
#define NUM_GRAPHS 64
#define NEG_SLOPE  0.2f

typedef __attribute__((ext_vector_type(16))) _Float16 v16h;
typedef __attribute__((ext_vector_type(8)))  _Float16 v8h;
typedef __attribute__((ext_vector_type(8)))  float    v8f;

// ---------------------------------------------------------------------------
// WMMA fragment helpers (wave32, v_wmma_f32_16x16x32_f16)
// A-matrix 16x32 per ISA 7.12.2: lane<16 -> M=lane,  K in {0..7, 16..23}
//                                lane>=16 -> M=lane-16, K in {8..15, 24..31}
// i.e. two contiguous 8-element K-runs per lane -> two 16B loads.
// B-matrix 32x16 (same K order as A): lane<16 holds K={0..7, 16..23},
// lane>=16 holds K={8..15, 24..31}, column N = lane%16 -> packed as 16 contiguous
// halves per lane.
// ---------------------------------------------------------------------------
__device__ __forceinline__ v16h load_a_frag(const _Float16* base, int stride,
                                            int lane, int ks) {
  int r  = lane & 15;
  int k0 = ks + ((lane < 16) ? 0 : 8);
  v8h lo = *(const v8h*)(base + r * stride + k0);
  v8h hi = *(const v8h*)(base + r * stride + k0 + 16);
  return __builtin_shufflevector(lo, hi, 0, 1, 2, 3, 4, 5, 6, 7,
                                 8, 9, 10, 11, 12, 13, 14, 15);
}
__device__ __forceinline__ v16h load_b_frag(const _Float16* __restrict__ pack,
                                            int tile, int lane) {
  return *(const v16h*)(pack + ((size_t)tile * 32 + lane) * 16);
}
__device__ __forceinline__ v8f wmma_f16(v16h a, v16h b, v8f c) {
  return __builtin_amdgcn_wmma_f32_16x16x32_f16(false, a, false, b, (short)0,
                                                c, false, false);
}

// ---------------- ordered-uint encoding for atomic float max ----------------
__device__ __forceinline__ unsigned f2ord(float f) {
  unsigned u = __float_as_uint(f);
  return (u & 0x80000000u) ? ~u : (u | 0x80000000u);
}
__device__ __forceinline__ float ord2f(unsigned e) {
  return (e & 0x80000000u) ? __uint_as_float(e & 0x7fffffffu)
                           : __uint_as_float(~e);
}
#define ORD_NEG_INF 0x007fffffu

__device__ __forceinline__ float leaky(float a) {
  return a > 0.f ? a : NEG_SLOPE * a;
}

// ---------------------------- init / pack kernels ---------------------------
__global__ void fill_u32_kernel(unsigned* __restrict__ p, unsigned v, int n) {
  int i = blockIdx.x * blockDim.x + threadIdx.x;
  if (i < n) p[i] = v;
}

__global__ void init_hcat_kernel(float* __restrict__ hcat,
                                 const float* __restrict__ b_i,
                                 const float* __restrict__ b_t) {
  int i = blockIdx.x * blockDim.x + threadIdx.x;
  if (i < N_NODES * 512) {
    int j = i & 511;
    hcat[i] = (j < 256) ? b_i[j] : b_t[j - 256];
  }
}

// x [N,20] f32 -> xp [N,32] f16, zero-padded K
__global__ void pack_x_kernel(const float* __restrict__ x,
                              _Float16* __restrict__ xp) {
  int t = blockIdx.x * blockDim.x + threadIdx.x;
  if (t >= N_NODES * 32) return;
  int node = t >> 5, k = t & 31;
  xp[t] = (k < 20) ? (_Float16)x[node * 20 + k] : (_Float16)0.f;
}

// W [K,N] f32 -> fragment-ordered f16: P[((kt*(N/16)+nt)*32+lane)*16+i]
//  = W[(kt*32 + (lane<16?0:8) + (i<8?i:i+8)) * N + nt*16 + lane%16]   (0 if k>=K)
__global__ void pack_w_kernel(const float* __restrict__ W,
                              _Float16* __restrict__ P, int K, int Kpad,
                              int N) {
  int t = blockIdx.x * blockDim.x + threadIdx.x;
  int total = (Kpad >> 5) * (N >> 4) * 512;
  if (t >= total) return;
  int i = t & 15;
  int lane = (t >> 4) & 31;
  int tile = t >> 9;
  int ntn = N >> 4;
  int nt = tile % ntn;
  int kt = tile / ntn;
  int k = kt * 32 + ((lane < 16) ? 0 : 8) + ((i < 8) ? i : i + 8);
  int col = nt * 16 + (lane & 15);
  P[t] = (k < K) ? (_Float16)W[k * N + col] : (_Float16)0.f;
}

// ------------------- h = x @ W  (20000x20 @ 20x256), WMMA -------------------
__global__ void __launch_bounds__(32)
feat_wmma_kernel(const _Float16* __restrict__ xp,
                 const _Float16* __restrict__ Wp, float* __restrict__ h) {
  const int lane = threadIdx.x;
  const int row0 = blockIdx.x * 16;
  const int col  = lane & 15;
  const int rsel = (lane >> 4) << 3;

  v16h a = load_a_frag(xp + (size_t)row0 * 32, 32, lane, 0);
  for (int nt = 0; nt < 16; ++nt) {
    v16h b = load_b_frag(Wp, nt, lane);
    v8f c = {};
    c = wmma_f16(a, b, c);
    float* out = h + (size_t)(row0 + rsel) * 256 + nt * 16 + col;
#pragma unroll
    for (int j = 0; j < 8; ++j) out[j * 256] = c[j];
  }
}

// ------------- per-(node,head) attention logits alpha_src/alpha_dst ---------
__global__ void alpha_kernel(const float* __restrict__ h,
                             const float* __restrict__ a_src,
                             const float* __restrict__ a_dst,
                             float* __restrict__ as, float* __restrict__ ad) {
  int t = blockIdx.x * blockDim.x + threadIdx.x;
  if (t >= N_NODES * H_HEADS) return;
  int node = t >> 4, head = t & 15;
  const float* hp = h + (size_t)node * 256 + head * 16;
  const float* sv = a_src + head * 16;
  const float* dv = a_dst + head * 16;
  float su = 0.f, du = 0.f;
#pragma unroll
  for (int c = 0; c < 16; ++c) {
    float v = hp[c];
    su += v * sv[c];
    du += v * dv[c];
  }
  as[t] = su;
  ad[t] = du;
}

// ----------------------------- edge passes ----------------------------------
__global__ void edge_max_kernel(const int* __restrict__ ei,
                                const float* __restrict__ as,
                                const float* __restrict__ ad,
                                unsigned* __restrict__ amax) {
  int t = blockIdx.x * blockDim.x + threadIdx.x;
  if (t >= N_EDGES * H_HEADS) return;
  int e = t >> 4, hd = t & 15;
  int s = ei[e], d = ei[N_EDGES + e];
  float a = leaky(as[s * 16 + hd] + ad[d * 16 + hd]);
  atomicMax(&amax[d * 16 + hd], f2ord(a));
}

__global__ void edge_denom_kernel(const int* __restrict__ ei,
                                  const float* __restrict__ as,
                                  const float* __restrict__ ad,
                                  const unsigned* __restrict__ amax,
                                  float* __restrict__ denom) {
  int t = blockIdx.x * blockDim.x + threadIdx.x;
  if (t >= N_EDGES * H_HEADS) return;
  int e = t >> 4, hd = t & 15;
  int s = ei[e], d = ei[N_EDGES + e];
  float a = leaky(as[s * 16 + hd] + ad[d * 16 + hd]);
  float ex = __expf(a - ord2f(amax[d * 16 + hd]));
  atomicAdd(&denom[d * 16 + hd], ex);
}

__global__ void edge_msg_kernel(const int* __restrict__ ei,
                                const float* __restrict__ as,
                                const float* __restrict__ ad,
                                const unsigned* __restrict__ amax,
                                const float* __restrict__ denom,
                                const float* __restrict__ h,
                                float* __restrict__ hcat, int colbase) {
  int t = blockIdx.x * blockDim.x + threadIdx.x;
  if (t >= N_EDGES * H_HEADS) return;
  int e = t >> 4, hd = t & 15;
  int s = ei[e], d = ei[N_EDGES + e];
  float a = leaky(as[s * 16 + hd] + ad[d * 16 + hd]);
  float dn = denom[d * 16 + hd];
  dn = (dn == 0.f) ? 1.f : dn;
  float attn = __expf(a - ord2f(amax[d * 16 + hd])) / dn;
  const float* hp = h + (size_t)s * 256 + hd * 16;
  float* op = hcat + (size_t)d * 512 + colbase + hd * 16;
#pragma unroll
  for (int c = 0; c < 16; ++c) atomicAdd(&op[c], attn * hp[c]);
}

// ------------- fused MLP: relu chain 512->64->32->16->16, all WMMA ----------
#define SAH 520   // f16 LDS row stride for the 16x512 tile (1040B, 16B-mult)
#define SIH 72    // stride for 16x64 intermediate (144B, 16B-mult)
#define SJH 40    // stride for 16x32 intermediates (80B, 16B-mult)
__global__ void __launch_bounds__(32)
mlp_wmma_kernel(const float* __restrict__ hcat,
                const _Float16* __restrict__ Wfp, const float* __restrict__ bf,
                const _Float16* __restrict__ W1p, const float* __restrict__ b1,
                const _Float16* __restrict__ W2p, const float* __restrict__ b2,
                const _Float16* __restrict__ W3p, const float* __restrict__ b3,
                float* __restrict__ h16) {
  __shared__ _Float16 sA[16 * SAH];
  __shared__ _Float16 sI[16 * SIH];
  const int lane = threadIdx.x;
  const int row0 = blockIdx.x * 16;
  const int col  = lane & 15;
  const int rsel = (lane >> 4) << 3;

  // stage hcat tile -> LDS f16 (vectorized f32x4 loads, full tiles only)
  for (int idx = lane * 4; idx < 16 * 512; idx += 32 * 4) {
    int r = idx >> 9, cl = idx & 511;
    const float4 v = *(const float4*)&hcat[(size_t)(row0 + r) * 512 + cl];
    sA[r * SAH + cl + 0] = (_Float16)v.x;
    sA[r * SAH + cl + 1] = (_Float16)v.y;
    sA[r * SAH + cl + 2] = (_Float16)v.z;
    sA[r * SAH + cl + 3] = (_Float16)v.w;
  }
  __syncthreads();

  // ---- layer 1: 512 -> 64 (16 K-steps x 4 N-tiles)
  v8f acc[4] = {};
  for (int kt = 0; kt < 16; ++kt) {
    if (kt + 1 < 16)
      __builtin_prefetch(Wfp + ((size_t)(kt + 1) * 4 * 32) * 16, 0, 1);
    v16h a = load_a_frag(sA, SAH, lane, kt * 32);
#pragma unroll
    for (int t = 0; t < 4; ++t) {
      v16h b = load_b_frag(Wfp, kt * 4 + t, lane);
      acc[t] = wmma_f16(a, b, acc[t]);
    }
  }
  __syncthreads();
#pragma unroll
  for (int t = 0; t < 4; ++t) {
    float bb = bf[t * 16 + col];
#pragma unroll
    for (int j = 0; j < 8; ++j)
      sI[(j + rsel) * SIH + t * 16 + col] =
          (_Float16)fmaxf(acc[t][j] + bb, 0.f);
  }
  __syncthreads();

  // ---- layer 2: 64 -> 32 (2 K-steps x 2 N-tiles)
  v8f a2[2] = {};
  for (int kt = 0; kt < 2; ++kt) {
    v16h a = load_a_frag(sI, SIH, lane, kt * 32);
#pragma unroll
    for (int t = 0; t < 2; ++t) {
      v16h b = load_b_frag(W1p, kt * 2 + t, lane);
      a2[t] = wmma_f16(a, b, a2[t]);
    }
  }
  __syncthreads();
#pragma unroll
  for (int t = 0; t < 2; ++t) {
    float bb = b1[t * 16 + col];
#pragma unroll
    for (int j = 0; j < 8; ++j)
      sI[(j + rsel) * SJH + t * 16 + col] =
          (_Float16)fmaxf(a2[t][j] + bb, 0.f);
  }
  __syncthreads();

  // ---- layer 3: 32 -> 16 (1 K-step)
  {
    v16h a = load_a_frag(sI, SJH, lane, 0);
    v16h b = load_b_frag(W2p, 0, lane);
    v8f a3 = {};
    a3 = wmma_f16(a, b, a3);
    __syncthreads();
    float bb = b2[col];
#pragma unroll
    for (int j = 0; j < 8; ++j) {
      sI[(j + rsel) * SJH + col]      = (_Float16)fmaxf(a3[j] + bb, 0.f);
      sI[(j + rsel) * SJH + 16 + col] = (_Float16)0.f;  // zero K-pad 16..31
    }
  }
  __syncthreads();

  // ---- layer 4: 16 -> 16 (K zero-padded to 32; W3p padded at pack time)
  {
    v16h a = load_a_frag(sI, SJH, lane, 0);
    v16h b = load_b_frag(W3p, 0, lane);
    v8f a4 = {};
    a4 = wmma_f16(a, b, a4);
    float bb = b3[col];
#pragma unroll
    for (int j = 0; j < 8; ++j)
      h16[(size_t)(row0 + j + rsel) * 16 + col] = fmaxf(a4[j] + bb, 0.f);
  }
}

// --------------------------- mean pooling -----------------------------------
__global__ void pool_kernel(const float* __restrict__ h16,
                            const int* __restrict__ batch,
                            float* __restrict__ ssum, float* __restrict__ cnt) {
  int t = blockIdx.x * blockDim.x + threadIdx.x;
  if (t >= N_NODES * 16) return;
  int node = t >> 4, c = t & 15;
  int g = batch[node];
  atomicAdd(&ssum[g * 16 + c], h16[t]);
  if (c == 0) atomicAdd(&cnt[g], 1.0f);
}

// ------------------------------ heads ---------------------------------------
__global__ void head_kernel(const float* __restrict__ ssum,
                            const float* __restrict__ cnt,
                            const float* __restrict__ Wce,
                            const float* __restrict__ bce,
                            const float* __restrict__ Wcv,
                            const float* __restrict__ bcv,
                            float* __restrict__ out) {
  int g = threadIdx.x;
  if (g >= NUM_GRAPHS) return;
  float c = cnt[g];
  float inv = (c == 0.f) ? 1.f : 1.f / c;
  float gv[16];
#pragma unroll
  for (int i = 0; i < 16; ++i) gv[i] = ssum[g * 16 + i] * inv;
#pragma unroll
  for (int o = 0; o < 3; ++o) {
    float se = bce[o], sv = bcv[o];
#pragma unroll
    for (int i = 0; i < 16; ++i) {
      se += gv[i] * Wce[i * 3 + o];
      sv += gv[i] * Wcv[i * 3 + o];
    }
    out[g * 3 + o] = se;
    out[NUM_GRAPHS * 3 + g * 3 + o] = sv;
  }
}

// ----------------------------- launcher -------------------------------------
extern "C" void kernel_launch(void* const* d_in, const int* in_sizes, int n_in,
                              void* d_out, int out_size, void* d_ws,
                              size_t ws_size, hipStream_t stream) {
  const float* x     = (const float*)d_in[0];
  const int*   ei_i  = (const int*)d_in[1];
  const int*   ei_t  = (const int*)d_in[3];
  const int*   batch = (const int*)d_in[5];
  const float* W_i   = (const float*)d_in[6];
  const float* aw_si = (const float*)d_in[7];
  const float* aw_di = (const float*)d_in[8];
  const float* b_i   = (const float*)d_in[9];
  const float* W_t   = (const float*)d_in[10];
  const float* aw_st = (const float*)d_in[11];
  const float* aw_dt = (const float*)d_in[12];
  const float* b_t   = (const float*)d_in[13];
  const float* Wf = (const float*)d_in[14]; const float* bf = (const float*)d_in[15];
  const float* W1 = (const float*)d_in[16]; const float* b1 = (const float*)d_in[17];
  const float* W2 = (const float*)d_in[18]; const float* b2 = (const float*)d_in[19];
  const float* W3 = (const float*)d_in[20]; const float* b3 = (const float*)d_in[21];
  const float* Wce = (const float*)d_in[22]; const float* bce = (const float*)d_in[23];
  const float* Wcv = (const float*)d_in[24]; const float* bcv = (const float*)d_in[25];
  float* out = (float*)d_out;

  // ---- workspace carve-up: float region, then f16 packed region
  float* ws = (float*)d_ws;
  size_t off = 0;
  float* h_i  = ws + off; off += (size_t)N_NODES * 256;
  float* h_t  = ws + off; off += (size_t)N_NODES * 256;
  float* hcat = ws + off; off += (size_t)N_NODES * 512;
  float* as_i = ws + off; off += (size_t)N_NODES * 16;
  float* ad_i = ws + off; off += (size_t)N_NODES * 16;
  float* as_t = ws + off; off += (size_t)N_NODES * 16;
  float* ad_t = ws + off; off += (size_t)N_NODES * 16;
  unsigned* amax_i = (unsigned*)(ws + off); off += (size_t)N_NODES * 16;
  float*    den_i  = ws + off;              off += (size_t)N_NODES * 16;
  unsigned* amax_t = (unsigned*)(ws + off); off += (size_t)N_NODES * 16;
  float*    den_t  = ws + off;              off += (size_t)N_NODES * 16;
  float* h16  = ws + off; off += (size_t)N_NODES * 16;
  float* ssum = ws + off; off += (size_t)NUM_GRAPHS * 16;
  float* cnt  = ws + off; off += (size_t)NUM_GRAPHS;

  _Float16* hp = (_Float16*)(ws + off);
  size_t hoff = 0;
  _Float16* xp  = hp + hoff; hoff += (size_t)N_NODES * 32;
  _Float16* Wip = hp + hoff; hoff += 1 * 16 * 512;   // K 20->32, N 256
  _Float16* Wtp = hp + hoff; hoff += 1 * 16 * 512;
  _Float16* Wfp = hp + hoff; hoff += 16 * 4 * 512;   // K 512, N 64
  _Float16* W1p = hp + hoff; hoff += 2 * 2 * 512;    // K 64,  N 32
  _Float16* W2p = hp + hoff; hoff += 1 * 1 * 512;    // K 32,  N 16
  _Float16* W3p = hp + hoff; hoff += 1 * 1 * 512;    // K 16->32, N 16

  const int TB = 256;
  const int nh = N_NODES * 16;
  const int eh = N_EDGES * 16;

  // ---- init + packing
  init_hcat_kernel<<<(N_NODES * 512 + TB - 1) / TB, TB, 0, stream>>>(hcat, b_i, b_t);
  fill_u32_kernel<<<(nh + TB - 1) / TB, TB, 0, stream>>>(amax_i, ORD_NEG_INF, nh);
  fill_u32_kernel<<<(nh + TB - 1) / TB, TB, 0, stream>>>(amax_t, ORD_NEG_INF, nh);
  fill_u32_kernel<<<(nh + TB - 1) / TB, TB, 0, stream>>>((unsigned*)den_i, 0u, nh);
  fill_u32_kernel<<<(nh + TB - 1) / TB, TB, 0, stream>>>((unsigned*)den_t, 0u, nh);
  fill_u32_kernel<<<1, TB, 0, stream>>>((unsigned*)ssum, 0u, NUM_GRAPHS * 16 + NUM_GRAPHS);

  pack_x_kernel<<<(N_NODES * 32 + TB - 1) / TB, TB, 0, stream>>>(x, xp);
  pack_w_kernel<<<(16 * 512 + TB - 1) / TB, TB, 0, stream>>>(W_i, Wip, 20, 32, 256);
  pack_w_kernel<<<(16 * 512 + TB - 1) / TB, TB, 0, stream>>>(W_t, Wtp, 20, 32, 256);
  pack_w_kernel<<<(64 * 512 + TB - 1) / TB, TB, 0, stream>>>(Wf, Wfp, 512, 512, 64);
  pack_w_kernel<<<(4 * 512 + TB - 1) / TB, TB, 0, stream>>>(W1, W1p, 64, 64, 32);
  pack_w_kernel<<<(1 * 512 + TB - 1) / TB, TB, 0, stream>>>(W2, W2p, 32, 32, 16);
  pack_w_kernel<<<(1 * 512 + TB - 1) / TB, TB, 0, stream>>>(W3, W3p, 16, 32, 16);

  // ---- feature transforms (WMMA)
  const int ntiles = N_NODES / 16;
  feat_wmma_kernel<<<ntiles, 32, 0, stream>>>(xp, Wip, h_i);
  feat_wmma_kernel<<<ntiles, 32, 0, stream>>>(xp, Wtp, h_t);

  // ---- attention logits
  alpha_kernel<<<(nh + TB - 1) / TB, TB, 0, stream>>>(h_i, aw_si, aw_di, as_i, ad_i);
  alpha_kernel<<<(nh + TB - 1) / TB, TB, 0, stream>>>(h_t, aw_st, aw_dt, as_t, ad_t);

  // ---- edge softmax + scatter
  int eb = (eh + TB - 1) / TB;
  edge_max_kernel<<<eb, TB, 0, stream>>>(ei_i, as_i, ad_i, amax_i);
  edge_max_kernel<<<eb, TB, 0, stream>>>(ei_t, as_t, ad_t, amax_t);
  edge_denom_kernel<<<eb, TB, 0, stream>>>(ei_i, as_i, ad_i, amax_i, den_i);
  edge_denom_kernel<<<eb, TB, 0, stream>>>(ei_t, as_t, ad_t, amax_t, den_t);
  edge_msg_kernel<<<eb, TB, 0, stream>>>(ei_i, as_i, ad_i, amax_i, den_i, h_i, hcat, 0);
  edge_msg_kernel<<<eb, TB, 0, stream>>>(ei_t, as_t, ad_t, amax_t, den_t, h_t, hcat, 256);

  // ---- fused MLP (WMMA)
  mlp_wmma_kernel<<<ntiles, 32, 0, stream>>>(hcat, Wfp, bf, W1p, b1, W2p, b2,
                                             W3p, b3, h16);

  // ---- pooling + heads
  pool_kernel<<<(nh + TB - 1) / TB, TB, 0, stream>>>(h16, batch, ssum, cnt);
  head_kernel<<<1, 64, 0, stream>>>(ssum, cnt, Wce, bce, Wcv, bcv, out);
}
